// DiTBlock_47442208752039
// MI455X (gfx1250) — matrix operand order repaired and checked by reading, not compiled, not run on hardware
//
#include <hip/hip_runtime.h>
#include <hip/hip_bf16.h>

// ---------------------------------------------------------------------------
// DiT block on MI455X (gfx1250).
//  - every GEMM runs on v_wmma_f32_16x16x32_bf16 with double-buffered async
//    LDS staging (global_load_async_to_lds_b128 / s_wait_asynccnt),
//    64-deep K stages (16 WMMAs per barrier pair)
//  - qkv GEMM fuses q/k rmsnorm + RoPE + head-layout scatter in its epilogue
//  - attention is one kernel: QK^T -> softmax (register stripe, shfl
//    reductions) -> P via LDS -> P@V partials -> ds_add_f32 reduce -> Ob bf16
//  - SwiGLU is a dual-B GEMM; proj / w2 GEMMs fuse bias+gate+residual
// ---------------------------------------------------------------------------

#define BN   4
#define NSEQ 1024
#define CC   1024
#define HH   16
#define DD   64
#define HID  2730
#define HIDP 2752            // HID padded to multiple of 64 for WMMA K/N dims
#define SIXC (6 * CC)
#define ROWS (BN * NSEQ)     // 4096
#define BH   (BN * HH)       // 64 attention heads total

#define MT 256               // block M tile
#define NT 64                // block N tile
#define KS 64                // staged K depth (2 WMMA k-steps per stage)

typedef __attribute__((ext_vector_type(16))) __bf16 v16bf;
typedef __attribute__((ext_vector_type(8)))  __bf16 v8bf;
typedef __attribute__((ext_vector_type(8)))  float  v8f;

static __device__ __forceinline__ __bf16 f2bf(float f) {
  unsigned u = __builtin_bit_cast(unsigned, f);
  unsigned r = (u + 0x7FFFu + ((u >> 16) & 1u)) >> 16;   // RNE
  unsigned short s = (unsigned short)r;
  return __builtin_bit_cast(__bf16, s);
}

// Generic pointer to __shared__ truncated to 32 bits == LDS byte offset.
static __device__ __forceinline__ unsigned lds_off(const void* p) {
  return (unsigned)(unsigned long long)p;
}

// Async copy 16 bytes global -> LDS (tracked by ASYNCcnt).
static __device__ __forceinline__ void async_copy16(unsigned lds, const void* g) {
  asm volatile("global_load_async_to_lds_b128 %0, %1, off"
               :: "v"(lds), "v"(g) : "memory");
}
static __device__ __forceinline__ void wait_async0() {
  asm volatile("s_wait_asynccnt 0" ::: "memory");
}

// ---------------------------------------------------------------------------
// Shared GEMM mainloop:  acc[2][4] (32x64 per wave) = A[M,K] @ Bt[N,K]^T tile.
// 256 thr = 8 waves; block tile 256x64; A(256x64)+B(64x64) double-buffered
// (80KB LDS); 16 WMMAs per stage between barriers.
// ---------------------------------------------------------------------------
static __device__ __forceinline__ void gemm_mainloop(
    const __bf16* __restrict__ A, const __bf16* __restrict__ Bt,
    int K, v8f acc[2][4]) {
  __shared__ __bf16 smem[2][MT * KS + NT * KS];   // 2 x 40KB

  const int tid  = threadIdx.x;
  const int wave = tid >> 5;
  const int lane = tid & 31;
  const int mBase = blockIdx.x * MT;
  const int n0    = blockIdx.y * NT;
  const int r  = lane & 15;
  const int hi = (lane >> 4) & 1;

  const int crow = tid >> 3;        // 0..31
  const int cseg = (tid & 7) * 8;   // element offset within a 64-elem row

  auto issue_copy = [&](int buf, int kk) {
    const __bf16* ga = A + (long)mBase * K + kk;
    __bf16* la = &smem[buf][0];
#pragma unroll
    for (int j = 0; j < 8; ++j) {
      int row = crow + j * 32;      // 0..255
      async_copy16(lds_off(&la[row * KS + cseg]), ga + (long)row * K + cseg);
    }
    const __bf16* gb = Bt + (long)n0 * K + kk;
    __bf16* lb = &smem[buf][MT * KS];
#pragma unroll
    for (int j = 0; j < 2; ++j) {
      int row = crow + j * 32;      // 0..63
      async_copy16(lds_off(&lb[row * KS + cseg]), gb + (long)row * K + cseg);
    }
  };

  issue_copy(0, 0);
  wait_async0();
  __syncthreads();

  int cur = 0;
  for (int kk = 0; kk < K; kk += KS, cur ^= 1) {
    if (kk + KS < K) issue_copy(cur ^ 1, kk + KS);

    const __bf16* abase = &smem[cur][0];
    const __bf16* bbase = &smem[cur][MT * KS];
#pragma unroll
    for (int half = 0; half < 2; ++half) {
      const int ko = half * 32;
      v16bf afr[2];
#pragma unroll
      for (int mi = 0; mi < 2; ++mi) {
        const __bf16* ap = abase + (wave * 32 + mi * 16 + r) * KS + ko + hi * 16;
        *(v8bf*)&afr[mi]         = *(const v8bf*)ap;
        *(((v8bf*)&afr[mi]) + 1) = *(const v8bf*)(ap + 8);
      }
      v16bf bfr[4];
#pragma unroll
      for (int t = 0; t < 4; ++t)
        bfr[t] = *(const v16bf*)(bbase + (t * 16 + r) * KS + ko + hi * 16);
#pragma unroll
      for (int t = 0; t < 4; ++t)
#pragma unroll
        for (int mi = 0; mi < 2; ++mi)
          acc[mi][t] = __builtin_amdgcn_wmma_f32_16x16x32_bf16(
              false, afr[mi], false, bfr[t], (short)0, acc[mi][t], false, false);
    }

    wait_async0();
    __syncthreads();
  }
}

// ---------------------------------------------------------------------------
// qkv GEMM with fused q/k rmsnorm + RoPE + head-layout scatter.
// Nn = 3C; each 64-col tile is exactly one (sel, head) D-chunk.
// q,k -> [BH, N, D] bf16 ;  v -> transposed [BH, D, N] bf16.
// ---------------------------------------------------------------------------
__global__ __launch_bounds__(256) void gemm_qkv(
    const __bf16* __restrict__ A, const __bf16* __restrict__ Bt,
    const float* __restrict__ qn_w, const float* __restrict__ kn_w,
    __bf16* __restrict__ qb, __bf16* __restrict__ kb,
    __bf16* __restrict__ vT, int K) {
  v8f acc[2][4] = {};
  gemm_mainloop(A, Bt, K, acc);

  const int wave = threadIdx.x >> 5;
  const int lane = threadIdx.x & 31;
  const int r  = lane & 15;
  const int hi = (lane >> 4) & 1;
  const int mBase = blockIdx.x * MT;
  const int n0    = blockIdx.y * NT;
  const int sel = n0 >> 10;              // 0=q, 1=k, 2=v
  const int h   = (n0 & 1023) >> 6;      // head
  const int b   = mBase >> 10;           // block-uniform batch (256 | 1024)
  const int z   = b * HH + h;

  if (sel == 2) {
#pragma unroll
    for (int mi = 0; mi < 2; ++mi) {
      const int row0 = mBase + wave * 32 + mi * 16 + (hi << 3);
#pragma unroll
      for (int t = 0; t < 4; ++t) {
        const int d = t * 16 + r;
#pragma unroll
        for (int i = 0; i < 8; ++i) {
          int n = (row0 + i) & (NSEQ - 1);
          vT[((long)z * DD + d) * NSEQ + n] = f2bf(acc[mi][t][i]);
        }
      }
    }
  } else {
    const float* w = sel ? kn_w : qn_w;
    __bf16* dst = sel ? kb : qb;
    float wv[4];
#pragma unroll
    for (int t = 0; t < 4; ++t) wv[t] = w[t * 16 + r];
    const float if0 = __powf(1.0e4f, -(float)(2 * r) * (1.f / DD));
    const float if1 = __powf(1.0e4f, -(float)(2 * (r + 16)) * (1.f / DD));
#pragma unroll
    for (int mi = 0; mi < 2; ++mi) {
      const int row0 = mBase + wave * 32 + mi * 16 + (hi << 3);
#pragma unroll
      for (int i = 0; i < 8; ++i) {
        float ss = 0.f;
#pragma unroll
        for (int t = 0; t < 4; ++t) ss += acc[mi][t][i] * acc[mi][t][i];
        for (int s = 1; s < 16; s <<= 1) ss += __shfl_xor(ss, s, 32);
        float rsq = rsqrtf(ss * (1.f / DD) + 1e-6f);
        float nv[4];
#pragma unroll
        for (int t = 0; t < 4; ++t) nv[t] = acc[mi][t][i] * rsq * wv[t];
        int n = (row0 + i) & (NSEQ - 1);
#pragma unroll
        for (int t = 0; t < 4; ++t) {
          float ang = (float)n * ((t & 1) ? if1 : if0);
          float cs = __cosf(ang), sn = __sinf(ang);
          float rot = (t < 2) ? -nv[t + 2] : nv[t - 2];
          dst[((long)z * NSEQ + n) * DD + t * 16 + r] = f2bf(nv[t] * cs + rot * sn);
        }
      }
    }
  }
}

// ---------------------------------------------------------------------------
// GEMM with fused  out = X + g[b, goff+c] * (A@Bt + bias[c])   (Nn = CC)
// ---------------------------------------------------------------------------
__global__ __launch_bounds__(256) void gemm_residual(
    const __bf16* __restrict__ A, const __bf16* __restrict__ Bt,
    const float* __restrict__ Xin, const float* __restrict__ bias, int has_bias,
    const float* __restrict__ mods, int goff, float* __restrict__ outF, int K) {
  v8f acc[2][4] = {};
  gemm_mainloop(A, Bt, K, acc);

  const int wave = threadIdx.x >> 5;
  const int lane = threadIdx.x & 31;
  const int r  = lane & 15;
  const int hi = (lane >> 4) & 1;
  const int mBase = blockIdx.x * MT;
  const int n0    = blockIdx.y * NT;
  const int b = mBase >> 10;

  float bval[4], gval[4];
#pragma unroll
  for (int t = 0; t < 4; ++t) {
    int cc = n0 + r + 16 * t;
    bval[t] = has_bias ? bias[cc] : 0.f;
    gval[t] = mods[b * SIXC + goff + cc];
  }
#pragma unroll
  for (int mi = 0; mi < 2; ++mi) {
    const int row0 = mBase + wave * 32 + mi * 16 + (hi << 3);
#pragma unroll
    for (int t = 0; t < 4; ++t) {
      const int cc = n0 + r + 16 * t;
#pragma unroll
      for (int i = 0; i < 8; ++i) {
        long idx = (long)(row0 + i) * CC + cc;
        outF[idx] = Xin[idx] + gval[t] * (acc[mi][t][i] + bval[t]);
      }
    }
  }
}

// ---------------------------------------------------------------------------
// Fused SwiGLU dual GEMM:  Hb = bf16( silu(A@B1) * (A@B3) )   [M, Nn] bf16
// ---------------------------------------------------------------------------
__global__ __launch_bounds__(256) void gemm_swiglu(
    const __bf16* __restrict__ A, const __bf16* __restrict__ B1t,
    const __bf16* __restrict__ B3t, __bf16* __restrict__ Hb,
    int Nn, int K) {
  __shared__ __bf16 smem[2][MT * KS + 2 * NT * KS];   // 2 x 48KB

  const int tid  = threadIdx.x;
  const int wave = tid >> 5;
  const int lane = tid & 31;
  const int mBase = blockIdx.x * MT;
  const int n0    = blockIdx.y * NT;
  const int r  = lane & 15;
  const int hi = (lane >> 4) & 1;

  const int crow = tid >> 3;
  const int cseg = (tid & 7) * 8;

  auto issue_copy = [&](int buf, int kk) {
    const __bf16* ga = A + (long)mBase * K + kk;
    __bf16* la = &smem[buf][0];
#pragma unroll
    for (int j = 0; j < 8; ++j) {
      int row = crow + j * 32;
      async_copy16(lds_off(&la[row * KS + cseg]), ga + (long)row * K + cseg);
    }
    const __bf16* g1 = B1t + (long)n0 * K + kk;
    const __bf16* g3 = B3t + (long)n0 * K + kk;
    __bf16* l1 = &smem[buf][MT * KS];
    __bf16* l3 = &smem[buf][MT * KS + NT * KS];
#pragma unroll
    for (int j = 0; j < 2; ++j) {
      int row = crow + j * 32;
      async_copy16(lds_off(&l1[row * KS + cseg]), g1 + (long)row * K + cseg);
      async_copy16(lds_off(&l3[row * KS + cseg]), g3 + (long)row * K + cseg);
    }
  };

  v8f acc1[2][4] = {};
  v8f acc3[2][4] = {};

  issue_copy(0, 0);
  wait_async0();
  __syncthreads();

  int cur = 0;
  for (int kk = 0; kk < K; kk += KS, cur ^= 1) {
    if (kk + KS < K) issue_copy(cur ^ 1, kk + KS);

    const __bf16* abase = &smem[cur][0];
    const __bf16* b1base = &smem[cur][MT * KS];
    const __bf16* b3base = &smem[cur][MT * KS + NT * KS];
#pragma unroll
    for (int half = 0; half < 2; ++half) {
      const int ko = half * 32;
      v16bf afr[2];
#pragma unroll
      for (int mi = 0; mi < 2; ++mi) {
        const __bf16* ap = abase + (wave * 32 + mi * 16 + r) * KS + ko + hi * 16;
        *(v8bf*)&afr[mi]         = *(const v8bf*)ap;
        *(((v8bf*)&afr[mi]) + 1) = *(const v8bf*)(ap + 8);
      }
      v16bf b1fr[4], b3fr[4];
#pragma unroll
      for (int t = 0; t < 4; ++t) {
        b1fr[t] = *(const v16bf*)(b1base + (t * 16 + r) * KS + ko + hi * 16);
        b3fr[t] = *(const v16bf*)(b3base + (t * 16 + r) * KS + ko + hi * 16);
      }
#pragma unroll
      for (int t = 0; t < 4; ++t)
#pragma unroll
        for (int mi = 0; mi < 2; ++mi) {
          acc1[mi][t] = __builtin_amdgcn_wmma_f32_16x16x32_bf16(
              false, afr[mi], false, b1fr[t], (short)0, acc1[mi][t], false, false);
          acc3[mi][t] = __builtin_amdgcn_wmma_f32_16x16x32_bf16(
              false, afr[mi], false, b3fr[t], (short)0, acc3[mi][t], false, false);
        }
    }

    wait_async0();
    __syncthreads();
  }

#pragma unroll
  for (int mi = 0; mi < 2; ++mi) {
    const int crow0 = mBase + wave * 32 + mi * 16 + (hi << 3);
#pragma unroll
    for (int t = 0; t < 4; ++t)
#pragma unroll
      for (int i = 0; i < 8; ++i) {
        float u = acc1[mi][t][i];
        float v = acc3[mi][t][i];
        Hb[(long)(crow0 + i) * Nn + n0 + r + 16 * t] =
            f2bf((u / (1.f + __expf(-u))) * v);
      }
  }
}

// ---------------------------------------------------------------------------
// Fully fused attention: Ob[b,n,h*D+d] = softmax(QK^T/8) @ V.
// Grid (NSEQ/16, 1, BH); 256 thr = 8 waves. 16-row stripe:
//  phase 1: S(16x1024) in registers (wave w holds cols [128w,128w+128))
//  phase 2: softmax via shfl_xor row reductions + tiny LDS cross-wave combine
//  phase 3: P bf16 -> LDS (32KB); each wave does P[:,128w:128w+128] @ V slice,
//           partial O(16x64) reduced with ds_add_f32; epilogue writes Ob bf16.
// ---------------------------------------------------------------------------
__global__ __launch_bounds__(256) void attn_fused(
    const __bf16* __restrict__ qb, const __bf16* __restrict__ kb,
    const __bf16* __restrict__ vT, __bf16* __restrict__ Ob) {
  const int z = blockIdx.z;
  const int bb = z >> 4;
  const int hh = z & 15;
  const int m0 = blockIdx.x * 16;
  const int tid  = threadIdx.x;
  const int wave = tid >> 5;
  const int lane = tid & 31;
  const int r  = lane & 15;
  const int hi = (lane >> 4) & 1;

  __shared__ __bf16 Plds[16][NSEQ];      // 32KB
  __shared__ float  Osum[16][DD];        // 4KB
  __shared__ float  redm[16][8];
  __shared__ float  reds[16][8];

  for (int i = tid; i < 16 * DD; i += 256) ((float*)Osum)[i] = 0.f;

  const __bf16* Q  = qb + (long)z * NSEQ * DD;
  const __bf16* Kb = kb + (long)z * NSEQ * DD;
  const __bf16* Vh = vT + (long)z * DD * NSEQ;

  // ---- phase 1: scores
  v16bf aq[2];
#pragma unroll
  for (int kt = 0; kt < 2; ++kt) {
    const __bf16* ap = Q + (long)(m0 + r) * DD + kt * 32 + hi * 16;
    *(v8bf*)&aq[kt]         = *(const v8bf*)ap;
    *(((v8bf*)&aq[kt]) + 1) = *(const v8bf*)(ap + 8);
  }
  v8f acc[2][4] = {};   // [nc][t]: this wave's 16x128 score chunk
#pragma unroll
  for (int nc = 0; nc < 2; ++nc) {
    const int n0 = wave * 128 + nc * 64;
#pragma unroll
    for (int kt = 0; kt < 2; ++kt)
#pragma unroll
      for (int t = 0; t < 4; ++t) {
        const __bf16* bp = Kb + (long)(n0 + t * 16 + r) * DD + kt * 32 + hi * 16;
        v16bf b = *(const v16bf*)bp;
        acc[nc][t] = __builtin_amdgcn_wmma_f32_16x16x32_bf16(
            false, aq[kt], false, b, (short)0, acc[nc][t], false, false);
      }
  }
#pragma unroll
  for (int nc = 0; nc < 2; ++nc)
#pragma unroll
    for (int t = 0; t < 4; ++t)
#pragma unroll
      for (int i = 0; i < 8; ++i) acc[nc][t][i] *= 0.125f;

  // ---- phase 2: softmax (row = hi*8 + i)
  float fmx[8];
#pragma unroll
  for (int i = 0; i < 8; ++i) {
    float m = acc[0][0][i];
#pragma unroll
    for (int nc = 0; nc < 2; ++nc)
#pragma unroll
      for (int t = 0; t < 4; ++t) m = fmaxf(m, acc[nc][t][i]);
    for (int s = 1; s < 16; s <<= 1) m = fmaxf(m, __shfl_xor(m, s, 32));
    fmx[i] = m;
  }
  if (r == 0)
#pragma unroll
    for (int i = 0; i < 8; ++i) redm[hi * 8 + i][wave] = fmx[i];
  __syncthreads();
#pragma unroll
  for (int i = 0; i < 8; ++i) {
    float m = redm[hi * 8 + i][0];
    for (int w = 1; w < 8; ++w) m = fmaxf(m, redm[hi * 8 + i][w]);
    fmx[i] = m;
  }
  float fsum[8];
#pragma unroll
  for (int i = 0; i < 8; ++i) {
    float s = 0.f;
#pragma unroll
    for (int nc = 0; nc < 2; ++nc)
#pragma unroll
      for (int t = 0; t < 4; ++t) {
        float e = __expf(acc[nc][t][i] - fmx[i]);
        acc[nc][t][i] = e;
        s += e;
      }
    for (int sh = 1; sh < 16; sh <<= 1) s += __shfl_xor(s, sh, 32);
    fsum[i] = s;
  }
  if (r == 0)
#pragma unroll
    for (int i = 0; i < 8; ++i) reds[hi * 8 + i][wave] = fsum[i];
  __syncthreads();
#pragma unroll
  for (int i = 0; i < 8; ++i) {
    float s = 0.f;
    for (int w = 0; w < 8; ++w) s += reds[hi * 8 + i][w];
    fsum[i] = 1.f / s;
  }

  // ---- P -> LDS (bf16)
#pragma unroll
  for (int nc = 0; nc < 2; ++nc)
#pragma unroll
    for (int t = 0; t < 4; ++t)
#pragma unroll
      for (int i = 0; i < 8; ++i)
        Plds[hi * 8 + i][wave * 128 + nc * 64 + t * 16 + r] =
            f2bf(acc[nc][t][i] * fsum[i]);
  __syncthreads();

  // ---- phase 3: O partial = P[:, 128w:128w+128] @ V-slice
  v8f oacc[4] = {};
#pragma unroll
  for (int kt = 0; kt < 4; ++kt) {
    const int kkw = wave * 128 + kt * 32;
    v16bf a;
    const __bf16* ap = &Plds[r][kkw + hi * 16];
    *(v8bf*)&a         = *(const v8bf*)ap;
    *(((v8bf*)&a) + 1) = *(const v8bf*)(ap + 8);
#pragma unroll
    for (int t = 0; t < 4; ++t) {
      const __bf16* bp = Vh + (long)(t * 16 + r) * NSEQ + kkw + hi * 16;
      v16bf b = *(const v16bf*)bp;
      oacc[t] = __builtin_amdgcn_wmma_f32_16x16x32_bf16(
          false, a, false, b, (short)0, oacc[t], false, false);
    }
  }
#pragma unroll
  for (int t = 0; t < 4; ++t)
#pragma unroll
    for (int i = 0; i < 8; ++i)
      atomicAdd(&Osum[hi * 8 + i][t * 16 + r], oacc[t][i]);
  __syncthreads();

  // ---- epilogue: write Ob in [B, N, H*D] layout
  for (int idx = tid; idx < 16 * DD; idx += 256) {
    int row = idx >> 6, d = idx & 63;
    int n = m0 + row;
    Ob[((long)(bb * NSEQ + n)) * CC + hh * DD + d] = f2bf(Osum[row][d]);
  }
}

// ---------------------------------------------------------------------------
__global__ void transpose_cast(const float* __restrict__ W, __bf16* __restrict__ Wt,
                               int Ksrc, int Nsrc, int Rows, int Cols) {
  long idx = (long)blockIdx.x * 256 + threadIdx.x;
  if (idx >= (long)Rows * Cols) return;
  int k = (int)(idx % Cols);
  int n = (int)(idx / Cols);
  float v = (n < Nsrc && k < Ksrc) ? W[(long)k * Nsrc + n] : 0.f;
  Wt[idx] = f2bf(v);
}

__global__ __launch_bounds__(256) void mods_kernel(
    const float* __restrict__ c, const float* __restrict__ ada_w,
    const float* __restrict__ ada_b, float* __restrict__ mods) {
  __shared__ float sc_[CC];
  int b = blockIdx.y;
  for (int i = threadIdx.x; i < CC; i += 256) {
    float v = c[b * CC + i];
    sc_[i] = v / (1.f + __expf(-v));
  }
  __syncthreads();
  int j = blockIdx.x * 256 + threadIdx.x;
  float acc = ada_b[j];
  for (int k = 0; k < CC; ++k) acc += sc_[k] * ada_w[(long)k * SIXC + j];
  mods[b * SIXC + j] = acc;
}

__global__ __launch_bounds__(256) void norm_mod_cast(
    const float* __restrict__ X, const float* __restrict__ mods,
    const float* __restrict__ nw, int sh_off, int sc_off,
    __bf16* __restrict__ out) {
  int row = blockIdx.x;
  int b = row >> 10;  // NSEQ == 1024
  const float* xr = X + (long)row * CC;
  __shared__ float red[256];
  float ss = 0.f;
  for (int i = threadIdx.x; i < CC; i += 256) { float v = xr[i]; ss += v * v; }
  red[threadIdx.x] = ss;
  __syncthreads();
  for (int s = 128; s > 0; s >>= 1) {
    if (threadIdx.x < s) red[threadIdx.x] += red[threadIdx.x + s];
    __syncthreads();
  }
  float rsq = rsqrtf(red[0] * (1.f / CC) + 1e-6f);
  const float* sh = mods + b * SIXC + sh_off;
  const float* sc = mods + b * SIXC + sc_off;
  for (int i = threadIdx.x; i < CC; i += 256) {
    float v = xr[i] * rsq * nw[i];
    out[(long)row * CC + i] = f2bf(v * (1.f + sc[i]) + sh[i]);
  }
}

// ---------------------------------------------------------------------------
extern "C" void kernel_launch(void* const* d_in, const int* in_sizes, int n_in,
                              void* d_out, int out_size, void* d_ws, size_t ws_size,
                              hipStream_t stream) {
  (void)in_sizes; (void)n_in; (void)out_size; (void)ws_size;
  const float* x       = (const float*)d_in[0];
  const float* c       = (const float*)d_in[1];
  const float* norm1_w = (const float*)d_in[2];
  const float* norm2_w = (const float*)d_in[3];
  const float* qkv_w   = (const float*)d_in[4];
  const float* proj_w  = (const float*)d_in[5];
  const float* proj_b  = (const float*)d_in[6];
  const float* qn_w    = (const float*)d_in[7];
  const float* kn_w    = (const float*)d_in[8];
  const float* w1      = (const float*)d_in[9];
  const float* w2      = (const float*)d_in[10];
  const float* w3      = (const float*)d_in[11];
  const float* ada_w   = (const float*)d_in[12];
  const float* ada_b   = (const float*)d_in[13];
  float* out = (float*)d_out;

  char* ws = (char*)d_ws;
  size_t off = 0;
  auto alloc = [&](size_t bytes) {
    size_t o = off;
    off += (bytes + 255) & ~(size_t)255;
    return o;
  };

  float*  modsb   = (float*) (ws + alloc((size_t)BN * SIXC * 4));
  __bf16* XM      = (__bf16*)(ws + alloc((size_t)ROWS * CC * 2));
  __bf16* qkv_wT  = (__bf16*)(ws + alloc((size_t)3 * CC * CC * 2));
  __bf16* qb      = (__bf16*)(ws + alloc((size_t)BH * NSEQ * DD * 2));
  __bf16* kb      = (__bf16*)(ws + alloc((size_t)BH * NSEQ * DD * 2));
  __bf16* vT      = (__bf16*)(ws + alloc((size_t)BH * DD * NSEQ * 2));
  __bf16* Ob      = (__bf16*)(ws + alloc((size_t)ROWS * CC * 2));
  __bf16* projT   = (__bf16*)(ws + alloc((size_t)CC * CC * 2));
  float*  x1      = (float*) (ws + alloc((size_t)ROWS * CC * 4));
  __bf16* XM2     = (__bf16*)(ws + alloc((size_t)ROWS * CC * 2));
  __bf16* w1T     = (__bf16*)(ws + alloc((size_t)HIDP * CC * 2));
  __bf16* w3T     = (__bf16*)(ws + alloc((size_t)HIDP * CC * 2));
  __bf16* w2T     = (__bf16*)(ws + alloc((size_t)CC * HIDP * 2));
  __bf16* Hb      = (__bf16*)(ws + alloc((size_t)ROWS * HIDP * 2));

  // --- weight casts/transposes (bf16, zero-padded) ---
  transpose_cast<<<(3 * CC * CC + 255) / 256, 256, 0, stream>>>(qkv_w, qkv_wT, CC, 3 * CC, 3 * CC, CC);
  transpose_cast<<<(CC * CC + 255) / 256, 256, 0, stream>>>(proj_w, projT, CC, CC, CC, CC);
  transpose_cast<<<(HIDP * CC + 255) / 256, 256, 0, stream>>>(w1, w1T, CC, HID, HIDP, CC);
  transpose_cast<<<(HIDP * CC + 255) / 256, 256, 0, stream>>>(w3, w3T, CC, HID, HIDP, CC);
  transpose_cast<<<(CC * HIDP + 255) / 256, 256, 0, stream>>>(w2, w2T, HID, CC, CC, HIDP);

  // --- adaLN modulation ---
  mods_kernel<<<dim3(SIXC / 256, BN), 256, 0, stream>>>(c, ada_w, ada_b, modsb);

  // --- attention branch ---
  norm_mod_cast<<<ROWS, 256, 0, stream>>>(x, modsb, norm1_w, 0, CC, XM);
  gemm_qkv<<<dim3(ROWS / MT, 3 * CC / NT, 1), 256, 0, stream>>>(
      XM, qkv_wT, qn_w, kn_w, qb, kb, vT, CC);
  attn_fused<<<dim3(NSEQ / 16, 1, BH), 256, 0, stream>>>(qb, kb, vT, Ob);
  gemm_residual<<<dim3(ROWS / MT, CC / NT, 1), 256, 0, stream>>>(
      Ob, projT, x, proj_b, 1, modsb, 2 * CC, x1, CC);

  // --- SwiGLU MLP branch ---
  norm_mod_cast<<<ROWS, 256, 0, stream>>>(x1, modsb, norm2_w, 3 * CC, 4 * CC, XM2);
  gemm_swiglu<<<dim3(ROWS / MT, HIDP / NT, 1), 256, 0, stream>>>(
      XM2, w1T, w3T, Hb, HIDP, CC);
  gemm_residual<<<dim3(ROWS / MT, CC / NT, 1), 256, 0, stream>>>(
      Hb, w2T, x1, (const float*)nullptr, 0, modsb, 5 * CC, out, HIDP);
}
